// Attention_86449101734039
// MI455X (gfx1250) — matrix operand order repaired and checked by reading, not compiled, not run on hardware
//
#include <hip/hip_runtime.h>
#include <hip/hip_bf16.h>

// MI455X (gfx1250) fused-attention pipeline, all GEMMs via v_wmma_f32_16x16x32_f16.
// B=4, S=2048, D=1024. Softmax over the QUERY axis (faithful to reference).
// Wave tile: 32x64 (2 A-frags x 4 B-frags -> 8 WMMAs per 32-wide K step).

typedef __attribute__((ext_vector_type(16))) _Float16 v16h;
typedef __attribute__((ext_vector_type(8)))  _Float16 v8h;
typedef __attribute__((ext_vector_type(8)))  float    v8f;
typedef __attribute__((ext_vector_type(4)))  float    v4f;

#define BB 4
#define SS 2048
#define DD 1024

static __device__ __forceinline__ v8f wmma16(v16h a, v16h b, v8f c) {
  // 8 args: (neg_a, A, neg_b, B, c_mod, C, reuse_a, reuse_b)
  return __builtin_amdgcn_wmma_f32_16x16x32_f16(false, a, false, b, (short)0, c,
                                                false, false);
}

// ---- Fragment loaders (wave32; lane m/n per CDNA5 ISA 7.12.2, but contiguous k order, not the ISA table order) ----
// A 16x32 f16: lane m=lane&15, h=lane>>4. elem e -> K=16h+e
//              (contiguous, matching B).
static __device__ __forceinline__ v16h load_a_f16(const _Float16* __restrict__ base,
                                                  int pitch, int m0, int k0, int lane) {
  int m = lane & 15, h = lane >> 4;
  const _Float16* p = base + (size_t)(m0 + m) * pitch + k0;
  v8h lo = *(const v8h*)(p + 16 * h);
  v8h hi = *(const v8h*)(p + 16 * h + 8);
  v16h a;
#pragma unroll
  for (int i = 0; i < 8; ++i) { a[i] = lo[i]; a[8 + i] = hi[i]; }
  return a;
}

static __device__ __forceinline__ v16h load_a_f32(const float* __restrict__ base,
                                                  int pitch, int m0, int k0, int lane) {
  int m = lane & 15, h = lane >> 4;
  const float* p = base + (size_t)(m0 + m) * pitch + k0;
  v4f l0 = *(const v4f*)(p + 16 * h);
  v4f l1 = *(const v4f*)(p + 16 * h + 4);
  v4f h0 = *(const v4f*)(p + 16 * h + 8);
  v4f h1 = *(const v4f*)(p + 16 * h + 12);
  v16h a;
#pragma unroll
  for (int i = 0; i < 4; ++i) {
    a[i]      = (_Float16)l0[i];
    a[4 + i]  = (_Float16)l1[i];
    a[8 + i]  = (_Float16)h0[i];
    a[12 + i] = (_Float16)h1[i];
  }
  return a;
}

// B 32x16 f16: element (k, n); source layout has column n contiguous along k:
// addr = base + n*pitch + k. lane n=lane&15, g=lane>>4, elem e -> K=16g+e.
static __device__ __forceinline__ v16h load_b_f16(const _Float16* __restrict__ base,
                                                  int pitch, int n0, int k0, int lane) {
  int n = lane & 15, g = lane >> 4;
  const _Float16* p = base + (size_t)(n0 + n) * pitch + k0 + 16 * g;
  v8h lo = *(const v8h*)(p);
  v8h hi = *(const v8h*)(p + 8);
  v16h b;
#pragma unroll
  for (int i = 0; i < 8; ++i) { b[i] = lo[i]; b[8 + i] = hi[i]; }
  return b;
}

static __device__ __forceinline__ v16h load_b_f32(const float* __restrict__ base,
                                                  int pitch, int n0, int k0, int lane) {
  int n = lane & 15, g = lane >> 4;
  const float* p = base + (size_t)(n0 + n) * pitch + k0 + 16 * g;
  v4f b0 = *(const v4f*)(p);
  v4f b1 = *(const v4f*)(p + 4);
  v4f b2 = *(const v4f*)(p + 8);
  v4f b3 = *(const v4f*)(p + 12);
  v16h b;
#pragma unroll
  for (int i = 0; i < 4; ++i) {
    b[i]      = (_Float16)b0[i];
    b[4 + i]  = (_Float16)b1[i];
    b[8 + i]  = (_Float16)b2[i];
    b[12 + i] = (_Float16)b3[i];
  }
  return b;
}

// ---- Kernel 1: projections. Out = X @ W^T  (W stored [out,in] row-major).
// A = X[m][k] (f32->f16), B element (k,n) = W[n][k] (contiguous along k).
// TRANS is a compile-time flag: straight-line store loops, no per-element branch.
template <int TRANS>
__global__ __launch_bounds__(128) void proj_kernel(
    const float* __restrict__ X,   // [B*S, D]
    const float* __restrict__ W,   // [D, D]
    _Float16* __restrict__ Out) {  // TRANS==0: [B*S, D], TRANS==1: [D, B*S]
  int lane = threadIdx.x & 31;
  int wave = threadIdx.x >> 5;
  int m0 = blockIdx.y * 64 + (wave >> 1) * 32;
  int n0 = blockIdx.x * 128 + (wave & 1) * 64;
  v8f acc[2][4] = {};
  for (int k0 = 0; k0 < DD; k0 += 32) {
    v16h a0 = load_a_f32(X, DD, m0,      k0, lane);
    v16h a1 = load_a_f32(X, DD, m0 + 16, k0, lane);
#pragma unroll
    for (int j = 0; j < 4; ++j) {
      v16h bf = load_b_f32(W, DD, n0 + 16 * j, k0, lane);
      acc[0][j] = wmma16(a0, bf, acc[0][j]);
      acc[1][j] = wmma16(a1, bf, acc[1][j]);
    }
  }
  int n = lane & 15, g = lane >> 4;
#pragma unroll
  for (int i = 0; i < 2; ++i)
#pragma unroll
    for (int j = 0; j < 4; ++j)
#pragma unroll
      for (int r = 0; r < 8; ++r) {
        int row = m0 + i * 16 + r + 8 * g;
        int col = n0 + j * 16 + n;
        _Float16 hv = (_Float16)acc[i][j][r];
        if (TRANS) Out[(size_t)col * (BB * SS) + row] = hv;  // V^T
        else       Out[(size_t)row * DD + col] = hv;
      }
}

// ---- Kernel 2: sims = (Q @ K^T) / sqrt(D), masked. B element (k,n)=K[n][k]. ----
__global__ __launch_bounds__(128) void scores_kernel(
    const _Float16* __restrict__ Q,        // [B, S, D]
    const _Float16* __restrict__ Kc,       // [B, S, D]
    const unsigned char* __restrict__ mask,// [S, S] bool
    float* __restrict__ sims) {            // [B, S, S]
  int b = blockIdx.z;
  const _Float16* Qb = Q + (size_t)b * SS * DD;
  const _Float16* Kb = Kc + (size_t)b * SS * DD;
  float* Sb = sims + (size_t)b * SS * SS;
  int lane = threadIdx.x & 31;
  int wave = threadIdx.x >> 5;
  int m0 = blockIdx.y * 64 + (wave >> 1) * 32;
  int n0 = blockIdx.x * 128 + (wave & 1) * 64;
  v8f acc[2][4] = {};
  for (int k0 = 0; k0 < DD; k0 += 32) {
    v16h a0 = load_a_f16(Qb, DD, m0,      k0, lane);
    v16h a1 = load_a_f16(Qb, DD, m0 + 16, k0, lane);
#pragma unroll
    for (int j = 0; j < 4; ++j) {
      v16h bf = load_b_f16(Kb, DD, n0 + 16 * j, k0, lane);
      acc[0][j] = wmma16(a0, bf, acc[0][j]);
      acc[1][j] = wmma16(a1, bf, acc[1][j]);
    }
  }
  const float scale = 0.03125f;  // 1/sqrt(1024)
  int n = lane & 15, g = lane >> 4;
#pragma unroll
  for (int i = 0; i < 2; ++i)
#pragma unroll
    for (int j = 0; j < 4; ++j)
#pragma unroll
      for (int r = 0; r < 8; ++r) {
        int q   = m0 + i * 16 + r + 8 * g;
        int col = n0 + j * 16 + n;
        float v = acc[i][j][r] * scale;
        if (mask[(size_t)q * SS + col]) v = -1e9f;
        Sb[(size_t)q * SS + col] = v;
      }
}

// ---- Kernel 3: softmax over the QUERY axis (axis=-2). One thread per key
// column; consecutive lanes read consecutive columns -> coalesced L2 lines. ----
__global__ __launch_bounds__(256) void softmax_q_kernel(float* __restrict__ sims) {
  int col = blockIdx.x * blockDim.x + threadIdx.x;   // key index
  int b   = blockIdx.y;
  float* base = sims + (size_t)b * SS * SS + col;
  float mx = -3.0e38f;
  for (int q = 0; q < SS; ++q) mx = fmaxf(mx, base[(size_t)q * SS]);
  float s = 0.0f;
  for (int q = 0; q < SS; ++q) {
    float e = __expf(base[(size_t)q * SS] - mx);
    base[(size_t)q * SS] = e;          // store numerator, avoid recomputing exp
    s += e;
  }
  float inv = 1.0f / s;
  for (int q = 0; q < SS; ++q)
    base[(size_t)q * SS] *= inv;
}

// ---- Kernel 4: out = attn @ V. A = attn (f32->f16), B element (k=key, n=d)
// = V^T[d][b*S+key] -> contiguous along key. ----
__global__ __launch_bounds__(128) void av_kernel(
    const float* __restrict__ attn,   // [B, S, S]
    const _Float16* __restrict__ Vt,  // [D, B*S]
    float* __restrict__ out) {        // [B, S, D]
  int b = blockIdx.z;
  const float* Ab = attn + (size_t)b * SS * SS;
  const _Float16* Vb = Vt + (size_t)b * SS;   // Vb[n*(B*S) + k] = V^T[d=n][b*S+k]
  int lane = threadIdx.x & 31;
  int wave = threadIdx.x >> 5;
  int m0 = blockIdx.y * 64 + (wave >> 1) * 32;    // query rows
  int n0 = blockIdx.x * 128 + (wave & 1) * 64;    // d columns
  v8f acc[2][4] = {};
  for (int k0 = 0; k0 < SS; k0 += 32) {
    v16h a0 = load_a_f32(Ab, SS, m0,      k0, lane);
    v16h a1 = load_a_f32(Ab, SS, m0 + 16, k0, lane);
#pragma unroll
    for (int j = 0; j < 4; ++j) {
      v16h bf = load_b_f16(Vb, BB * SS, n0 + 16 * j, k0, lane);
      acc[0][j] = wmma16(a0, bf, acc[0][j]);
      acc[1][j] = wmma16(a1, bf, acc[1][j]);
    }
  }
  int n = lane & 15, g = lane >> 4;
  float* Ob = out + (size_t)b * SS * DD;
#pragma unroll
  for (int i = 0; i < 2; ++i)
#pragma unroll
    for (int j = 0; j < 4; ++j)
#pragma unroll
      for (int r = 0; r < 8; ++r) {
        int row = m0 + i * 16 + r + 8 * g;
        int col = n0 + j * 16 + n;
        Ob[(size_t)row * DD + col] = acc[i][j][r];
      }
}

extern "C" void kernel_launch(void* const* d_in, const int* in_sizes, int n_in,
                              void* d_out, int out_size, void* d_ws, size_t ws_size,
                              hipStream_t stream) {
  const float* xq = (const float*)d_in[0];
  const float* xk = (const float*)d_in[1];
  const float* xv = (const float*)d_in[2];
  const unsigned char* mask = (const unsigned char*)d_in[3];
  const float* Wq = (const float*)d_in[4];
  const float* Wk = (const float*)d_in[5];
  const float* Wv = (const float*)d_in[6];

  // Workspace layout (112 MB total; resides in the 192 MB L2):
  //   [0, 16MB)   Q   f16 [B,S,D]
  //   [16, 32MB)  K   f16 [B,S,D]
  //   [32, 48MB)  V^T f16 [D, B*S]
  //   [48, 112MB) sims/attn f32 [B,S,S]
  char* ws = (char*)d_ws;
  _Float16* Q  = (_Float16*)(ws);
  _Float16* Km = (_Float16*)(ws + ((size_t)16 << 20));
  _Float16* Vt = (_Float16*)(ws + ((size_t)32 << 20));
  float*  sims = (float*)   (ws + ((size_t)48 << 20));

  dim3 blk(128);
  dim3 gproj(DD / 128, (BB * SS) / 64);
  proj_kernel<0><<<gproj, blk, 0, stream>>>(xq, Wq, Q);
  proj_kernel<0><<<gproj, blk, 0, stream>>>(xk, Wk, Km);
  proj_kernel<1><<<gproj, blk, 0, stream>>>(xv, Wv, Vt);

  dim3 gsc(SS / 128, SS / 64, BB);
  scores_kernel<<<gsc, blk, 0, stream>>>(Q, Km, mask, sims);

  softmax_q_kernel<<<dim3(SS / 256, BB), 256, 0, stream>>>(sims);

  dim3 gav(DD / 128, SS / 64, BB);
  av_kernel<<<gav, blk, 0, stream>>>(sims, Vt, (float*)d_out);
}
